// myYOLO_13211319403107
// MI455X (gfx1250) — matrix operand order repaired and checked by reading, not compiled, not run on hardware
//
#include <hip/hip_runtime.h>
#include <hip/hip_bf16.h>

typedef __attribute__((ext_vector_type(16))) _Float16 v16h;
typedef __attribute__((ext_vector_type(8)))  float    v8f;

#define BATCH 8
#define HS 14
#define WS 14
#define NCELLS 196
#define NCLS 20
#define HEAD_OUT 25

// ---------------------------------------------------------------------------
// Layer 1: direct fp32 conv 3->64, stride 2, SAME(pad 0,1), ReLU -> NHWC f16
// ---------------------------------------------------------------------------
__global__ void conv1_kernel(const float* __restrict__ x,
                             const float* __restrict__ w1,
                             _Float16* __restrict__ act1)
{
    const int total = BATCH * 224 * 224 * 64;
    int tid = blockIdx.x * blockDim.x + threadIdx.x;
    if (tid >= total) return;
    int co = tid & 63;
    int t  = tid >> 6;
    int ox = t % 224; t /= 224;
    int oy = t % 224; t /= 224;
    int b  = t;
    float s = 0.f;
    #pragma unroll
    for (int ci = 0; ci < 3; ++ci) {
        #pragma unroll
        for (int ky = 0; ky < 3; ++ky) {
            int iy = 2 * oy + ky;
            if (iy >= 448) continue;
            #pragma unroll
            for (int kx = 0; kx < 3; ++kx) {
                int ix = 2 * ox + kx;
                if (ix >= 448) continue;
                s += x[((size_t)(b * 3 + ci) * 448 + iy) * 448 + ix] *
                     w1[((co * 3 + ci) * 3 + ky) * 3 + kx];
            }
        }
    }
    s = s > 0.f ? s : 0.f;
    act1[tid] = (_Float16)s;   // tid == ((b*224+oy)*224+ox)*64+co  (NHWC)
}

// ---------------------------------------------------------------------------
// Weight repack: OIHW f32 -> OHWI f16  (k = (ky*3+kx)*Cin + ci)
// ---------------------------------------------------------------------------
__global__ void cvtw_kernel(const float* __restrict__ src,
                            _Float16* __restrict__ dst, int Cin, int total)
{
    int i = blockIdx.x * blockDim.x + threadIdx.x;
    if (i >= total) return;
    int K  = Cin * 9;
    int co = i / K;
    int r  = i - co * K;
    int p  = r / Cin;          // ky*3+kx
    int ci = r - p * Cin;
    dst[i] = (_Float16)src[(size_t)co * K + (size_t)ci * 9 + p];
}

// ---------------------------------------------------------------------------
// Implicit-GEMM conv, stride 2, 3x3, SAME(0,1), ReLU.
// One wave32 computes a 32(Cout) x 32(spatial) block = 2x2 WMMA tiles with
// 4 f32 accumulators; A fragments reused across N-tiles, B across M-tiles.
// K is ordered (tap p, channel ci); CIN is a compile-time multiple of 32 so
// the channel loop is division-free with constant strides.
// Activations NHWC f16, weights OHWI f16 -> all fragment loads aligned b32.
// ---------------------------------------------------------------------------
template<int CIN>
__global__ void conv_wmma_kernel(const _Float16* __restrict__ actIn,
                                 const _Float16* __restrict__ w,
                                 _Float16* __restrict__ actOut,
                                 int Cout, int Hi, int Wi,
                                 int Ho, int Wo, int Ntot)
{
    const int lane = threadIdx.x;    // 0..31
    const int half = lane >> 4;
    const int lc   = lane & 15;
    const int K    = CIN * 9;

    const int nBase = blockIdx.x * 32;   // 2 spatial tiles
    const int mBase = blockIdx.y * 32;   // 2 Cout tiles

    // spatial decode for the two N tiles (n = nBase + nt*16 + lc)
    int b_[2], oy_[2], ox_[2];
    bool nv_[2];
    const int hw = Ho * Wo;
    #pragma unroll
    for (int nt = 0; nt < 2; ++nt) {
        int n = nBase + nt * 16 + lc;
        nv_[nt] = (n < Ntot);
        if (!nv_[nt]) n = Ntot - 1;
        int b  = n / hw;
        int rr = n - b * hw;
        b_[nt]  = b;
        oy_[nt] = rr / Wo;
        ox_[nt] = rr - oy_[nt] * Wo;
    }

    const _Float16* wRow0 = w + (size_t)(mBase + lc) * K;       // A tile 0
    const _Float16* wRow1 = w + (size_t)(mBase + 16 + lc) * K;  // A tile 1

    union Frag { unsigned u[8]; v16h h; };
    v8f a00 = {}, a01 = {}, a10 = {}, a11 = {};

    #pragma unroll
    for (int p = 0; p < 9; ++p) {
        const int ky = p / 3, kx = p - (p / 3) * 3;
        // per-tap base pointers + validity for both N tiles
        const _Float16* bp[2];
        bool bv[2];
        #pragma unroll
        for (int nt = 0; nt < 2; ++nt) {
            int iy = 2 * oy_[nt] + ky;
            int ix = 2 * ox_[nt] + kx;
            bv[nt] = (iy < Hi) && (ix < Wi);
            bp[nt] = actIn + ((size_t)(b_[nt] * Hi + iy) * Wi + ix) * CIN;
        }
        const int kb = p * CIN;

        #pragma unroll 4
        for (int c0 = 0; c0 < CIN; c0 += 32) {
            Frag A0, A1, B0, B1;
            // A 16x32 f16: VGPR j holds K pair at (j>>2)*16 + half*8 + (j&3)*2
            #pragma unroll
            for (int j = 0; j < 8; ++j) {
                int o = kb + c0 + ((j >> 2) << 4) + (half << 3) + ((j & 3) << 1);
                A0.u[j] = *(const unsigned*)(wRow0 + o);
                A1.u[j] = *(const unsigned*)(wRow1 + o);
            }
            // B 32x16 f16: N = lane&15, VGPR j holds K pair at (j>>2)*16 + half*8 + (j&3)*2
            #pragma unroll
            for (int j = 0; j < 8; ++j) {
                int o = c0 + ((j >> 2) << 4) + (half << 3) + ((j & 3) << 1);
                B0.u[j] = bv[0] ? *(const unsigned*)(bp[0] + o) : 0u;
                B1.u[j] = bv[1] ? *(const unsigned*)(bp[1] + o) : 0u;
            }
            a00 = __builtin_amdgcn_wmma_f32_16x16x32_f16(false, A0.h, false, B0.h,
                                                         (short)0, a00, false, false);
            a01 = __builtin_amdgcn_wmma_f32_16x16x32_f16(false, A0.h, false, B1.h,
                                                         (short)0, a01, false, false);
            a10 = __builtin_amdgcn_wmma_f32_16x16x32_f16(false, A1.h, false, B0.h,
                                                         (short)0, a10, false, false);
            a11 = __builtin_amdgcn_wmma_f32_16x16x32_f16(false, A1.h, false, B1.h,
                                                         (short)0, a11, false, false);
        }
    }

    // D f32 16x16: value (mt,nt,r) at Cout = mBase+mt*16+half*8+r, spatial n-tile nt
    #pragma unroll
    for (int nt = 0; nt < 2; ++nt) {
        if (!nv_[nt]) continue;
        size_t outBase =
            ((size_t)(b_[nt] * Ho + oy_[nt]) * Wo + ox_[nt]) * Cout + mBase;
        const v8f* accs[2] = { nt == 0 ? &a00 : &a01, nt == 0 ? &a10 : &a11 };
        #pragma unroll
        for (int mt = 0; mt < 2; ++mt) {
            #pragma unroll
            for (int r = 0; r < 8; ++r) {
                float v = (*accs[mt])[r];
                v = v > 0.f ? v : 0.f;
                actOut[outBase + mt * 16 + (half << 3) + r] = (_Float16)v;
            }
        }
    }
}

// ---------------------------------------------------------------------------
// Head: 1x1 conv 512->25, one wave per (b,cell), lane o<25 dots 512
// ---------------------------------------------------------------------------
__global__ void head_kernel(const _Float16* __restrict__ act5,
                            const float* __restrict__ wh,
                            float* __restrict__ pred)
{
    int cellb = blockIdx.x;            // 0 .. B*NCELLS-1
    int o = threadIdx.x;
    if (o >= HEAD_OUT) return;
    const _Float16* a = act5 + (size_t)cellb * 512;
    const float*    w = wh + (size_t)o * 512;
    float s = 0.f;
    for (int c = 0; c < 512; ++c) s += (float)a[c] * w[c];
    pred[(size_t)cellb * HEAD_OUT + o] = s;
}

// ---------------------------------------------------------------------------
// Decode: sigmoid conf, softmax*conf, box decode+clip
// ---------------------------------------------------------------------------
__global__ void decode_kernel(const float* __restrict__ pred,
                              float* __restrict__ cls,
                              float* __restrict__ boxes,
                              float* __restrict__ out)
{
    int i = blockIdx.x * blockDim.x + threadIdx.x;   // 0 .. B*NCELLS-1
    if (i >= BATCH * NCELLS) return;
    const float* p = pred + (size_t)i * HEAD_OUT;
    float conf = 1.f / (1.f + expf(-p[0]));
    float mx = p[1];
    for (int c = 1; c < NCLS; ++c) mx = fmaxf(mx, p[1 + c]);
    float e[NCLS], sum = 0.f;
    for (int c = 0; c < NCLS; ++c) { e[c] = expf(p[1 + c] - mx); sum += e[c]; }
    for (int c = 0; c < NCLS; ++c)
        cls[(size_t)i * NCLS + c] = e[c] / sum * conf;

    int cell = i % NCELLS;
    float gx = (float)(cell % WS), gy = (float)(cell / WS);
    float xc = (1.f / (1.f + expf(-p[21])) + gx) * 32.f;
    float yc = (1.f / (1.f + expf(-p[22])) + gy) * 32.f;
    float bw = expf(p[23]) * 32.f;
    float bh = expf(p[24]) * 32.f;
    float x1 = fminf(fmaxf((xc - bw * 0.5f) / 448.f, 0.f), 1.f);
    float y1 = fminf(fmaxf((yc - bh * 0.5f) / 448.f, 0.f), 1.f);
    float x2 = fminf(fmaxf((xc + bw * 0.5f) / 448.f, 0.f), 1.f);
    float y2 = fminf(fmaxf((yc + bh * 0.5f) / 448.f, 0.f), 1.f);
    boxes[(size_t)i * 4 + 0] = x1;  boxes[(size_t)i * 4 + 1] = y1;
    boxes[(size_t)i * 4 + 2] = x2;  boxes[(size_t)i * 4 + 3] = y2;
    out[(size_t)i * 24 + 0] = x1;   out[(size_t)i * 24 + 1] = y1;
    out[(size_t)i * 24 + 2] = x2;   out[(size_t)i * 24 + 3] = y2;
}

// ---------------------------------------------------------------------------
// Per-batch 196x196 IoU matrix
// ---------------------------------------------------------------------------
__global__ void iou_kernel(const float* __restrict__ boxes,
                           float* __restrict__ iou)
{
    const int total = BATCH * NCELLS * NCELLS;
    int t = blockIdx.x * blockDim.x + threadIdx.x;
    if (t >= total) return;
    int j = t % NCELLS; int u = t / NCELLS;
    int i = u % NCELLS; int b = u / NCELLS;
    const float* bi = boxes + ((size_t)b * NCELLS + i) * 4;
    const float* bj = boxes + ((size_t)b * NCELLS + j) * 4;
    float ai = fmaxf(bi[2] - bi[0], 0.f) * fmaxf(bi[3] - bi[1], 0.f);
    float aj = fmaxf(bj[2] - bj[0], 0.f) * fmaxf(bj[3] - bj[1], 0.f);
    float ix1 = fmaxf(bi[0], bj[0]), iy1 = fmaxf(bi[1], bj[1]);
    float ix2 = fminf(bi[2], bj[2]), iy2 = fminf(bi[3], bj[3]);
    float inter = fmaxf(ix2 - ix1, 0.f) * fmaxf(iy2 - iy1, 0.f);
    iou[t] = inter / (ai + aj - inter + 1e-14f);
}

// ---------------------------------------------------------------------------
// Per (batch,class) greedy NMS; stable argsort via rank counting
// ---------------------------------------------------------------------------
__global__ void nms_kernel(const float* __restrict__ cls,
                           const float* __restrict__ iou,
                           float* __restrict__ out)
{
    __shared__ float s[NCELLS];
    __shared__ short order[NCELLS];
    __shared__ unsigned char keep[NCELLS];
    int b = blockIdx.x / NCLS;
    int c = blockIdx.x % NCLS;
    int tid = threadIdx.x;

    if (tid < NCELLS) s[tid] = cls[((size_t)b * NCELLS + tid) * NCLS + c];
    __syncthreads();
    if (tid < NCELLS) {
        float si = s[tid];
        int rank = 0;
        for (int j = 0; j < NCELLS; ++j) {
            float sj = s[j];
            rank += (sj > si) || (sj == si && j < tid);
        }
        order[rank] = (short)tid;
    }
    __syncthreads();
    if (tid < NCELLS) keep[tid] = (s[order[tid]] > 0.01f) ? 1 : 0;
    __syncthreads();
    for (int i = 0; i < NCELLS; ++i) {
        if (tid < NCELLS && tid > i && keep[i]) {
            const float* row = iou + ((size_t)b * NCELLS + order[i]) * NCELLS;
            if (row[order[tid]] > 0.5f) keep[tid] = 0;
        }
        __syncthreads();
    }
    if (tid < NCELLS) {
        int cell = order[tid];
        out[((size_t)b * NCELLS + cell) * 24 + 4 + c] = keep[tid] ? s[cell] : 0.f;
    }
}

// ---------------------------------------------------------------------------
extern "C" void kernel_launch(void* const* d_in, const int* in_sizes, int n_in,
                              void* d_out, int out_size, void* d_ws, size_t ws_size,
                              hipStream_t stream)
{
    const float* x  = (const float*)d_in[0];
    const float* w1 = (const float*)d_in[1];
    const float* w2 = (const float*)d_in[2];
    const float* w3 = (const float*)d_in[3];
    const float* w4 = (const float*)d_in[4];
    const float* w5 = (const float*)d_in[5];
    const float* wh = (const float*)d_in[6];
    float* out = (float*)d_out;

    char* ws = (char*)d_ws;
    size_t off = 0;
    auto alloc = [&](size_t bytes) -> void* {
        void* p = ws + off;
        off = (off + bytes + 255) & ~(size_t)255;
        return p;
    };

    _Float16* wf2  = (_Float16*)alloc((size_t)128 * 576  * 2);
    _Float16* wf3  = (_Float16*)alloc((size_t)256 * 1152 * 2);
    _Float16* wf4  = (_Float16*)alloc((size_t)512 * 2304 * 2);
    _Float16* wf5  = (_Float16*)alloc((size_t)512 * 4608 * 2);
    _Float16* act1 = (_Float16*)alloc((size_t)BATCH * 224 * 224 * 64  * 2);
    _Float16* act2 = (_Float16*)alloc((size_t)BATCH * 112 * 112 * 128 * 2);
    _Float16* act3 = (_Float16*)alloc((size_t)BATCH * 56 * 56 * 256 * 2);
    _Float16* act4 = (_Float16*)alloc((size_t)BATCH * 28 * 28 * 512 * 2);
    _Float16* act5 = (_Float16*)alloc((size_t)BATCH * 14 * 14 * 512 * 2);
    float* pred  = (float*)alloc((size_t)BATCH * NCELLS * HEAD_OUT * 4);
    float* clsS  = (float*)alloc((size_t)BATCH * NCELLS * NCLS * 4);
    float* boxes = (float*)alloc((size_t)BATCH * NCELLS * 4 * 4);
    float* iouM  = (float*)alloc((size_t)BATCH * NCELLS * NCELLS * 4);

    // weight repack (f32 OIHW -> f16 OHWI)
    cvtw_kernel<<<(128 * 576  + 255) / 256, 256, 0, stream>>>(w2, wf2, 64,  128 * 576);
    cvtw_kernel<<<(256 * 1152 + 255) / 256, 256, 0, stream>>>(w3, wf3, 128, 256 * 1152);
    cvtw_kernel<<<(512 * 2304 + 255) / 256, 256, 0, stream>>>(w4, wf4, 256, 512 * 2304);
    cvtw_kernel<<<(512 * 4608 + 255) / 256, 256, 0, stream>>>(w5, wf5, 512, 512 * 4608);

    // layer 1 (fp32 direct)
    {
        int total = BATCH * 224 * 224 * 64;
        conv1_kernel<<<(total + 255) / 256, 256, 0, stream>>>(x, w1, act1);
    }

    // layers 2..5 : WMMA implicit GEMM, 32x32 blocks per wave
    {
        int N = BATCH * 112 * 112;   // 100352
        conv_wmma_kernel<64><<<dim3((N + 31) / 32, 128 / 32), 32, 0, stream>>>(
            act1, wf2, act2, 128, 224, 224, 112, 112, N);
    }
    {
        int N = BATCH * 56 * 56;     // 25088
        conv_wmma_kernel<128><<<dim3((N + 31) / 32, 256 / 32), 32, 0, stream>>>(
            act2, wf3, act3, 256, 112, 112, 56, 56, N);
    }
    {
        int N = BATCH * 28 * 28;     // 6272
        conv_wmma_kernel<256><<<dim3((N + 31) / 32, 512 / 32), 32, 0, stream>>>(
            act3, wf4, act4, 512, 56, 56, 28, 28, N);
    }
    {
        int N = BATCH * 14 * 14;     // 1568
        conv_wmma_kernel<512><<<dim3((N + 31) / 32, 512 / 32), 32, 0, stream>>>(
            act4, wf5, act5, 512, 28, 28, 14, 14, N);
    }

    // head + decode + iou + nms
    head_kernel<<<BATCH * NCELLS, 32, 0, stream>>>(act5, wh, pred);
    decode_kernel<<<(BATCH * NCELLS + 255) / 256, 256, 0, stream>>>(pred, clsS, boxes, out);
    iou_kernel<<<(BATCH * NCELLS * NCELLS + 255) / 256, 256, 0, stream>>>(boxes, iouM);
    nms_kernel<<<BATCH * NCLS, 256, 0, stream>>>(clsS, iouM, out);
}
